// SymbolicRerankerV2_32057635897353
// MI455X (gfx1250) — compile-verified, matrix operand order checked by reading, not run on hardware
//
#include <hip/hip_runtime.h>
#include <hip/hip_bf16.h>
#include <math.h>

typedef __attribute__((ext_vector_type(16))) _Float16 v16h;
typedef __attribute__((ext_vector_type(8)))  float    v8f;

#define B_    8192
#define C_    3755
#define R_    214
#define S_    13
#define SCB_  30
#define ST_   6
#define H_    64
#define K_    20

// ---------------------------------------------------------------------------
// Block reduction helpers (256 threads = 8 waves of 32 on gfx1250 wave32)
// ---------------------------------------------------------------------------
__device__ __forceinline__ float waveSum(float v) {
#pragma unroll
    for (int m = 16; m >= 1; m >>= 1) v += __shfl_xor(v, m);
    return v;
}
__device__ __forceinline__ float waveMax(float v) {
#pragma unroll
    for (int m = 16; m >= 1; m >>= 1) v = fmaxf(v, __shfl_xor(v, m));
    return v;
}
__device__ __forceinline__ float blockSum(float v, float* red, int t) {
    v = waveSum(v);
    __syncthreads();
    if ((t & 31) == 0) red[t >> 5] = v;
    __syncthreads();
    float s = 0.f;
#pragma unroll
    for (int i = 0; i < 8; ++i) s += red[i];
    return s;
}
__device__ __forceinline__ float blockMax(float v, float* red, int t) {
    v = waveMax(v);
    __syncthreads();
    if ((t & 31) == 0) red[t >> 5] = v;
    __syncthreads();
    float s = -INFINITY;
#pragma unroll
    for (int i = 0; i < 8; ++i) s = fmaxf(s, red[i]);
    return s;
}

// ---------------------------------------------------------------------------
// Kernel 1: one block per batch row. Fuses the char_logits copy with
// softmax stats, top-20 selection, and all 6 feature computations.
// Row (15KB) lives in LDS (320KB/WGP available).
// ---------------------------------------------------------------------------
__global__ __launch_bounds__(256) void row_kernel(
    const float* __restrict__ charL,  const float* __restrict__ radL,
    const float* __restrict__ structL,const float* __restrict__ scL,
    const float* __restrict__ styL,
    const int*   __restrict__ radMask,const int* __restrict__ structLab,
    const int*   __restrict__ scLab,  const float* __restrict__ sig,
    float* __restrict__ out,
    float* __restrict__ gTopV, int* __restrict__ gTopI,
    float* __restrict__ gFeat)
{
    __shared__ float sC[C_];        // char row (destroyed during top-k)
    __shared__ float sR[R_];        // radical sigmoid probs
    __shared__ float sSP[S_];       // structure softmax probs
    __shared__ float sPN[ST_];      // normalized stroke-type prediction
    __shared__ float sTopV[K_];
    __shared__ int   sTopI[K_];
    __shared__ float redV[8];
    __shared__ int   redI[8];
    __shared__ float sStroke;       // argmax(stroke_count) as float

    const int b = blockIdx.x;
    const int t = threadIdx.x;
    const float* row  = charL + (size_t)b * C_;
    float*       orow = out   + (size_t)b * C_;

    // ---- phase 1: load row to LDS, fused copy to output, local max ----
    float lmax = -INFINITY;
    for (int i = t; i < C_; i += 256) {
        float v = row[i];
        sC[i]   = v;
        orow[i] = v;                 // fused pass-through copy
        lmax = fmaxf(lmax, v);
    }
    __syncthreads();
    const float rowmax = blockMax(lmax, redV, t);

    // ---- phase 2: softmax denominator over full row ----
    float ls = 0.f;
    for (int i = t; i < C_; i += 256) ls += expf(sC[i] - rowmax);
    const float denom = blockSum(ls, redV, t);

    // ---- phase 3: radical sigmoid probs + total mass ----
    float rsum = 0.f;
    for (int i = t; i < R_; i += 256) {
        float p = 1.0f / (1.0f + expf(-radL[(size_t)b * R_ + i]));
        sR[i] = p;
        rsum += p;
    }
    const float totalRaw = blockSum(rsum, redV, t);
    const float totalP   = fmaxf(totalRaw, 1e-6f);

    // ---- phase 4: tiny per-row scalars (13/30/6 wide) on thread 0 ----
    if (t == 0) {
        // structure softmax (S=13)
        float sv[S_]; float m = -INFINITY;
        for (int s = 0; s < S_; ++s) { sv[s] = structL[(size_t)b * S_ + s]; m = fmaxf(m, sv[s]); }
        float ssum = 0.f;
        for (int s = 0; s < S_; ++s) { float e = expf(sv[s] - m); sSP[s] = e; ssum += e; }
        for (int s = 0; s < S_; ++s) sSP[s] /= ssum;
        // stroke-count argmax (SCB=30), lowest index on tie
        int   am = 0; float bv = scL[(size_t)b * SCB_];
        for (int j = 1; j < SCB_; ++j) {
            float v = scL[(size_t)b * SCB_ + j];
            if (v > bv) { bv = v; am = j; }
        }
        sStroke = (float)am;
        // normalized stroke-type prediction (ST=6)
        float v6[ST_]; float nn = 0.f;
        for (int j = 0; j < ST_; ++j) { v6[j] = styL[(size_t)b * ST_ + j]; nn += v6[j] * v6[j]; }
        float d = fmaxf(sqrtf(nn), 1e-12f);
        for (int j = 0; j < ST_; ++j) sPN[j] = v6[j] / d;
    }
    __syncthreads();

    // ---- phase 5: iterative top-20 (argmax + mask to -inf) ----
    for (int it = 0; it < K_; ++it) {
        float bv = -INFINITY; int bi = 0x7fffffff;
        for (int i = t; i < C_; i += 256) {
            float v = sC[i];
            if (v > bv) { bv = v; bi = i; }
        }
#pragma unroll
        for (int m = 16; m >= 1; m >>= 1) {
            float ov = __shfl_xor(bv, m);
            int   oi = __shfl_xor(bi, m);
            if (ov > bv || (ov == bv && oi < bi)) { bv = ov; bi = oi; }
        }
        if ((t & 31) == 0) { redV[t >> 5] = bv; redI[t >> 5] = bi; }
        __syncthreads();
        if (t == 0) {
            float fv = redV[0]; int fi = redI[0];
            for (int w = 1; w < 8; ++w)
                if (redV[w] > fv || (redV[w] == fv && redI[w] < fi)) { fv = redV[w]; fi = redI[w]; }
            sTopV[it] = fv; sTopI[it] = fi;
            sC[fi] = -INFINITY;
        }
        __syncthreads();
    }

    // ---- phase 6: features, one candidate per wave ----
    const int wave = t >> 5, lane = t & 31;
    for (int k = wave; k < K_; k += 8) {
        const int cand = sTopI[k];
        const int* mrow = radMask + (size_t)cand * R_;
        float det = 0.f, cnt = 0.f;
        for (int r = lane; r < R_; r += 32) {
            float mv = (float)mrow[r];
            det += sR[r] * mv;
            cnt += mv;
        }
        det = waveSum(det);
        cnt = waveSum(cnt);
        if (lane == 0) {
            float rmr = det / fmaxf(cnt, 1.f);
            float rfr = (totalRaw - det) / totalP;
            float sm  = sSP[structLab[cand]];
            int   sp  = (int)sStroke;
            float sd  = fabsf((float)(sp - scLab[cand])) * (1.0f / 29.0f);
            // zero-masked stroke-type cosine
            float cs[ST_]; float nv = 0.f;
            for (int j = 0; j < ST_; ++j) { cs[j] = sig[(size_t)cand * ST_ + j]; nv += cs[j] * cs[j]; }
            nv = sqrtf(nv);
            float has = (nv > 1e-6f) ? 1.f : 0.f;
            float an = 0.f; float adj[ST_];
            for (int j = 0; j < ST_; ++j) { adj[j] = cs[j] + 1e-8f * (1.f - has); an += adj[j] * adj[j]; }
            float d2 = fmaxf(sqrtf(an), 1e-12f);
            float cosv = 0.f;
            for (int j = 0; j < ST_; ++j) cosv += sPN[j] * (adj[j] / d2);
            cosv *= has;
            float conf = expf(sTopV[k] - rowmax) / denom;
            size_t base = ((size_t)b * K_ + k) * 6;
            gFeat[base + 0] = rmr;
            gFeat[base + 1] = rfr;
            gFeat[base + 2] = sm;
            gFeat[base + 3] = sd;
            gFeat[base + 4] = cosv;
            gFeat[base + 5] = conf;
        }
    }
    if (t < K_) {
        gTopV[(size_t)b * K_ + t] = sTopV[t];
        gTopI[(size_t)b * K_ + t] = sTopI[t];
    }
}

// ---------------------------------------------------------------------------
// Kernel 2: reranker MLP with WMMA + scatter.
// Each wave handles 16 (b,k) rows: A = 16x32 f16 (K=0..5 valid, rest zero),
// B tiles = W1 columns (4 N-tiles of 16), f32 accum, then bias+ReLU+W2 dot
// folded per-lane using the documented C layout, shfl-xor reduced per half.
// ---------------------------------------------------------------------------
__global__ __launch_bounds__(256) void mlp_kernel(
    const float* __restrict__ gTopV, const int* __restrict__ gTopI,
    const float* __restrict__ gFeat,
    const float* __restrict__ W1, const float* __restrict__ b1,
    const float* __restrict__ W2, const float* __restrict__ b2,
    const float* __restrict__ rw,
    float* __restrict__ out)
{
    const int t = threadIdx.x;
    const int wave = t >> 5, lane = t & 31;
    const int rowBase = (blockIdx.x * 8 + wave) * 16;   // 16 (b,k) rows per wave

    // A tile: 16-bit A 16x32 layout -> lanes 0-15 hold K=0..7 in elements 0..7
    // (VGPR0..3); only K=0..5 are real, everything else zero. Lanes 16-31
    // carry K=8..15 / K=24..31 -> all zero here.
    v16h a = {};
    if (lane < 16) {
        const float* f = gFeat + (size_t)(rowBase + lane) * 6;
#pragma unroll
        for (int j = 0; j < 6; ++j) a[j] = (_Float16)f[j];
    }

    float accs[8];
#pragma unroll
    for (int r = 0; r < 8; ++r) accs[r] = 0.f;

#pragma unroll
    for (int tile = 0; tile < 4; ++tile) {
        // B tile: 32x16, lane = column N, elements = K rows; K=0..5 valid.
        v16h bm = {};
        if (lane < 16) {
            const int n = tile * 16 + lane;
#pragma unroll
            for (int j = 0; j < 6; ++j) bm[j] = (_Float16)W1[j * H_ + n];
        }
        v8f c = {};
        c = __builtin_amdgcn_wmma_f32_16x16x32_f16(
                /*neg_a=*/false, a, /*neg_b=*/false, bm,
                /*c_mod=*/(short)0, c, /*reuse_a=*/false, /*reuse_b=*/false);
        // C layout: lane l, VGPR r holds (M = r + 8*(l>=16), N = l%16 + 16*tile)
        const int n   = tile * 16 + (lane & 15);
        const float bn = b1[n];
        const float wn = W2[n];
#pragma unroll
        for (int r = 0; r < 8; ++r) {
            float h = c[r] + bn;
            h = fmaxf(h, 0.f);
            accs[r] += h * wn;
        }
    }
    // Reduce across the 16 lanes of each half (masks < 16 stay within halves)
#pragma unroll
    for (int m = 8; m >= 1; m >>= 1) {
#pragma unroll
        for (int r = 0; r < 8; ++r) accs[r] += __shfl_xor(accs[r], m);
    }

    const float b2v = b2[0];
    const float rwv = rw[0];
    const int half = lane >> 4;        // 0: M=0..7, 1: M=8..15
    const int j    = lane & 15;
    if (j < 8) {
        const int m      = j + 8 * half;
        const int rowIdx = rowBase + m;
        const float score = accs[j] + b2v;
        const float comb  = gTopV[rowIdx] + rwv * score;
        const int bb = rowIdx / K_;
        out[(size_t)bb * C_ + gTopI[rowIdx]] = comb;
    }
}

// ---------------------------------------------------------------------------
extern "C" void kernel_launch(void* const* d_in, const int* in_sizes, int n_in,
                              void* d_out, int out_size, void* d_ws, size_t ws_size,
                              hipStream_t stream) {
    (void)in_sizes; (void)n_in; (void)out_size; (void)ws_size;
    const float* charL    = (const float*)d_in[0];
    const float* radL     = (const float*)d_in[1];
    const float* structL  = (const float*)d_in[2];
    const float* scL      = (const float*)d_in[3];
    const float* styL     = (const float*)d_in[4];
    const int*   radMask  = (const int*)  d_in[5];
    const int*   structLab= (const int*)  d_in[6];
    const int*   scLab    = (const int*)  d_in[7];
    const float* sig      = (const float*)d_in[8];
    const float* W1       = (const float*)d_in[9];
    const float* b1       = (const float*)d_in[10];
    const float* W2       = (const float*)d_in[11];
    const float* b2       = (const float*)d_in[12];
    const float* rw       = (const float*)d_in[13];
    float* out = (float*)d_out;

    // workspace layout
    char* ws = (char*)d_ws;
    float* gTopV = (float*)ws;                                   // B*K floats
    int*   gTopI = (int*)(ws + (size_t)B_ * K_ * sizeof(float)); // B*K ints
    float* gFeat = (float*)(ws + 2u * (size_t)B_ * K_ * sizeof(float)); // B*K*6

    row_kernel<<<B_, 256, 0, stream>>>(charL, radL, structL, scL, styL,
                                       radMask, structLab, scLab, sig,
                                       out, gTopV, gTopI, gFeat);

    // (B*K)/16 waves, 8 waves/block -> B*K/128 blocks
    mlp_kernel<<<(B_ * K_) / 128, 256, 0, stream>>>(gTopV, gTopI, gFeat,
                                                    W1, b1, W2, b2, rw, out);
}
